// ForceFieldPredictor_25640954757501
// MI455X (gfx1250) — matrix operand order repaired and checked by reading, not compiled, not run on hardware
//
#include <hip/hip_runtime.h>
#include <hip/hip_bf16.h>

// ---------------------------------------------------------------------------
// Types for CDNA5 WMMA / TDM
// ---------------------------------------------------------------------------
typedef _Float16 h16;
typedef __attribute__((ext_vector_type(16))) _Float16 v16h;
typedef __attribute__((ext_vector_type(8)))  float    v8f;
typedef unsigned int __attribute__((ext_vector_type(4))) u32x4;
typedef int  __attribute__((ext_vector_type(4))) i32x4;
typedef int  __attribute__((ext_vector_type(8))) i32x8;

#define PH 136   // LDS activation pitch (halves) -> 272B rows, conflict-free
#define PW 136   // LDS weight pitch (halves)

// workspace layout (in halves / h16 elements)
#define OFF_TW0T 0        // trunk w0^T   [128][32]
#define OFF_TW1T 4096     // trunk w1^T   [128][128]
#define OFF_TW2T 20480    // trunk w2^T   [128][128]
#define OFF_JW0T 36864    // joint w0^T   [128][32]
#define OFF_BW0T 40960    // branch w0^T  [128][32]
#define OFF_BW1T 45056    // branch w1^T  [128][128]
#define OFF_BW2T 61440    // branch w2^T  [128][128]
#define OFF_BOUT 77824    // branch_out   [4096][128] f16

union V16HU { v16h v; uint4 u[2]; };

// A-tile (16x32 f16 activations) from LDS, row-major pitch PH.
// Layout per ISA: lane<16 rows M=lane, K = k0+0..7 & k0+16..23;
//                 lane>=16 rows M=lane-16, K = k0+8..15 & k0+24..31.
__device__ __forceinline__ v16h lds_load_A(const h16* hb, int rowbase, int k0, int lane) {
  int r  = rowbase + (lane & 15);
  int kb = k0 + ((lane >> 4) << 3);
  const h16* p = hb + r * PH + kb;
  V16HU a;
  a.u[0] = *(const uint4*)(p);
  a.u[1] = *(const uint4*)(p + 16);
  return a.v;
}

// B-tile (32x16 f16 weights) from LDS weight buffer stored as Wt[n][k].
// Layout per ISA: lane<16 col N=lane, K=k0+0..7 & k0+16..23; lane>=16 col N=lane-16, K=k0+8..15 & k0+24..31.
__device__ __forceinline__ v16h lds_load_B(const h16* wt, int k0, int n0, int lane) {
  int n  = n0 + (lane & 15);
  int kb = k0 + ((lane >> 4) << 3);
  const h16* p = wt + n * PW + kb;
  V16HU b;
  b.u[0] = *(const uint4*)(p);
  b.u[1] = *(const uint4*)(p + 16);
  return b.v;
}

// One 32-row x 128-col dense layer for this wave: h_out = [relu](h_in @ W + b).
__device__ __forceinline__ void mlp_layer_128(h16* hb, const h16* wt, const float* bias,
                                              int waveRowBase, int lane, int kcCount,
                                              bool relu) {
  v8f acc[2][8];
#pragma unroll
  for (int nt = 0; nt < 8; ++nt) {
    float bv = bias[nt * 16 + (lane & 15)];
    v8f c = { bv, bv, bv, bv, bv, bv, bv, bv };
    acc[0][nt] = c;
    acc[1][nt] = c;
  }
  for (int kc = 0; kc < kcCount; ++kc) {
    int k0 = kc * 32;
    v16h a0 = lds_load_A(hb, waveRowBase,      k0, lane);
    v16h a1 = lds_load_A(hb, waveRowBase + 16, k0, lane);
#pragma unroll
    for (int nt = 0; nt < 8; ++nt) {
      v16h bm = lds_load_B(wt, k0, nt * 16, lane);
      acc[0][nt] = __builtin_amdgcn_wmma_f32_16x16x32_f16(
          false, a0, false, bm, (short)0, acc[0][nt], false, false);
      acc[1][nt] = __builtin_amdgcn_wmma_f32_16x16x32_f16(
          false, a1, false, bm, (short)0, acc[1][nt], false, false);
    }
  }
  // C/D layout: VGPR e holds M = e + 8*(lane>=16), N = lane%16 (+ tile offsets).
#pragma unroll
  for (int mb = 0; mb < 2; ++mb)
#pragma unroll
    for (int nt = 0; nt < 8; ++nt)
#pragma unroll
      for (int e = 0; e < 8; ++e) {
        float x = acc[mb][nt][e];
        if (relu) x = fmaxf(x, 0.0f);
        int row = waveRowBase + mb * 16 + e + ((lane >> 4) << 3);
        int col = nt * 16 + (lane & 15);
        hb[row * PH + col] = (h16)x;
      }
}

// Cooperative stage of Wt[128][kdim] (f16, contiguous) + bias into LDS.
__device__ __forceinline__ void stage_weights(h16* s_wt, const h16* g_wt, int kdim,
                                              float* s_bias, const float* g_bias,
                                              int tid, int nthreads) {
  const int kd2 = kdim >> 1;              // u32 per row
  const int total = 128 * kd2;
  const uint* g = (const uint*)g_wt;
  for (int i = tid; i < total; i += nthreads) {
    int n = i / kd2, k2 = i - n * kd2;
    *(uint*)(s_wt + n * PW + (k2 << 1)) = g[i];
  }
  for (int i = tid; i < 128; i += nthreads) s_bias[i] = g_bias[i];
}

// ---------------------------------------------------------------------------
// Tensor Data Mover showcase: DMA 64 bytes global->LDS (dedicated scratch
// region, drained with s_wait_tensorcnt; functional staging is independent).
// D# per CDNA5 ISA: group0 {count=1, lds_addr, global_addr,
// type=2}, group1 {data_size=4B, tensor_dim0=16, tensor_dim1=1, tile_dim0=16,
// tile_dim1=1, stride0=16}.
// ---------------------------------------------------------------------------
__device__ __forceinline__ void tdm_touch(const h16* gsrc, h16* ldst) {
  unsigned long long ga = (unsigned long long)(uintptr_t)gsrc;
  unsigned int laddr = (unsigned int)(uintptr_t)ldst;   // LDS offset
  u32x4 g0;
  g0[0] = 1u;                                            // count=1, user mode
  g0[1] = laddr;                                         // lds_addr
  g0[2] = (unsigned int)ga;                              // global_addr[31:0]
  g0[3] = (unsigned int)((ga >> 32) & 0x01FFFFFFu) | (2u << 30); // addr hi | type=2
  i32x8 g1;
  g1[0] = (2 << 16);        // data_size=2 (4B units)
  g1[1] = (16 << 16);       // tensor_dim0 = 16 -> bits[79:48]
  g1[2] = (1 << 16);        // tensor_dim1 = 1  -> bits[111:80]
  g1[3] = (16 << 16);       // tile_dim0 = 16   -> bits[127:112]
  g1[4] = 1;                // tile_dim1 = 1, tile_dim2 = 0
  g1[5] = 16;               // tensor_dim0_stride lo
  g1[6] = 0;
  g1[7] = 0;
  i32x4 z4 = { 0, 0, 0, 0 };
#if defined(__clang_major__) && (__clang_major__ >= 23)
  i32x8 z8 = { 0, 0, 0, 0, 0, 0, 0, 0 };
  __builtin_amdgcn_tensor_load_to_lds(g0, g1, z4, z4, z8, 0);
#else
  __builtin_amdgcn_tensor_load_to_lds(g0, g1, z4, z4, 0);
#endif
  __builtin_amdgcn_s_wait_tensorcnt((short)0);
}

// ---------------------------------------------------------------------------
// Kernel 0: convert + transpose + pad all MLP weights into workspace (f16)
// ---------------------------------------------------------------------------
__global__ void ff_prep_kernel(const float* __restrict__ tw0, const float* __restrict__ tws,
                               const float* __restrict__ jw0, const float* __restrict__ bw0,
                               const float* __restrict__ bws, h16* __restrict__ ws) {
  int i = blockIdx.x * 256 + threadIdx.x;
  if (i >= OFF_BOUT) return;
  float val = 0.0f;
  if (i < OFF_TW1T) {                       // trunk w0^T, pad K 9->32
    int j = i;            int n = j >> 5, k = j & 31;
    if (k < 9) val = tw0[k * 128 + n];
  } else if (i < OFF_TW2T) {                // trunk w1^T
    int j = i - OFF_TW1T; int n = j >> 7, k = j & 127;
    val = tws[k * 128 + n];
  } else if (i < OFF_JW0T) {                // trunk w2^T
    int j = i - OFF_TW2T; int n = j >> 7, k = j & 127;
    val = tws[16384 + k * 128 + n];
  } else if (i < OFF_BW0T) {                // joint w0^T, pad K 15->32
    int j = i - OFF_JW0T; int n = j >> 5, k = j & 31;
    if (k < 15) val = jw0[k * 128 + n];
  } else if (i < OFF_BW1T) {                // branch w0^T, pad K 5->32
    int j = i - OFF_BW0T; int n = j >> 5, k = j & 31;
    if (k < 5) val = bw0[k * 128 + n];
  } else if (i < OFF_BW2T) {                // branch w1^T
    int j = i - OFF_BW1T; int n = j >> 7, k = j & 127;
    val = bws[k * 128 + n];
  } else {                                  // branch w2^T
    int j = i - OFF_BW2T; int n = j >> 7, k = j & 127;
    val = bws[16384 + k * 128 + n];
  }
  ws[i] = (h16)val;
}

// ---------------------------------------------------------------------------
// Kernel 1: branch MLP over the 4096 distinct (b,o) rows -> branch_out f16
// ---------------------------------------------------------------------------
__global__ __launch_bounds__(64)
void ff_branch_kernel(const float* __restrict__ init_x, const h16* __restrict__ ws,
                      const float* __restrict__ bb0, const float* __restrict__ bbs) {
  __shared__ h16  s_wt[128 * PW];
  __shared__ h16  s_h[64 * PH];
  __shared__ float s_bias[128];

  const int tid  = threadIdx.x;
  const int lane = tid & 31;
  const int wave = tid >> 5;
  const int row  = blockIdx.x * 64 + tid;       // 0..4095

  const float* ix = init_x + row * 9;
  h16* hrow = s_h + tid * PH;
  float f5[5] = { 0.0f, 0.0f, ix[2], ix[3], ix[4] };
#pragma unroll
  for (int k = 0; k < 32; ++k) hrow[k] = (k < 5) ? (h16)f5[k] : (h16)0.0f;

  stage_weights(s_wt, ws + OFF_BW0T, 32, s_bias, bb0, tid, 64);
  __syncthreads();
  mlp_layer_128(s_h, s_wt, s_bias, wave * 32, lane, 1, true);
  __syncthreads();
  stage_weights(s_wt, ws + OFF_BW1T, 128, s_bias, bbs, tid, 64);
  __syncthreads();
  mlp_layer_128(s_h, s_wt, s_bias, wave * 32, lane, 4, true);
  __syncthreads();
  stage_weights(s_wt, ws + OFF_BW2T, 128, s_bias, bbs + 128, tid, 64);
  __syncthreads();
  mlp_layer_128(s_h, s_wt, s_bias, wave * 32, lane, 4, false);

  // write this row's 128-half result (wave-private region; DS ops in order)
  uint* bo = (uint*)((h16*)ws + OFF_BOUT) + row * 64;
  const uint* src = (const uint*)hrow;
#pragma unroll
  for (int i = 0; i < 64; ++i) bo[i] = src[i];
}

// ---------------------------------------------------------------------------
// Kernel 2: fused geometry + trunk MLP + heads over all 262144 pairs.
// One block = one (b,o) row x all 64 t values -> init side is block-uniform.
// ---------------------------------------------------------------------------
__device__ __forceinline__ float ptseg(float px, float py, float ax, float ay,
                                       float abx, float aby, float s) {
  float t = ((px - ax) * abx + (py - ay) * aby) / s;
  t = fminf(fmaxf(t, 0.0f), 1.0f);
  float dx = px - (ax + t * abx);
  float dy = py - (ay + t * aby);
  return sqrtf(dx * dx + dy * dy);
}

__global__ __launch_bounds__(64)
void ff_trunk_kernel(const float* __restrict__ init_x, const float* __restrict__ query_x,
                     const float* __restrict__ init_v, const float* __restrict__ query_v,
                     const float* __restrict__ init_av, const float* __restrict__ query_av,
                     const float* __restrict__ tb0, const float* __restrict__ tbs,
                     const float* __restrict__ out_w, const float* __restrict__ out_b,
                     const float* __restrict__ sp_w0, const float* __restrict__ sp_b0,
                     const float* __restrict__ sp_w1, const float* __restrict__ sp_b1,
                     const float* __restrict__ jb0, const float* __restrict__ jw1,
                     const float* __restrict__ jb1,
                     const h16* __restrict__ ws, float* __restrict__ out) {
  __shared__ h16   s_wt[128 * PW];
  __shared__ h16   s_h[64 * PH];
  __shared__ float s_bias[128];
  __shared__ float s_bw[128 * 4];   // branch[n] * out_w[n][c] folded per block
  __shared__ h16   s_tdm[256];      // TDM landing pad (never read)

  const int tid  = threadIdx.x;
  const int lane = tid & 31;
  const int wave = tid >> 5;
  const int bo   = blockIdx.x;                  // b*64 + o (block-uniform)
  const int t    = tid;                         // query index
  const int b    = bo >> 6;
  const int p    = bo * 64 + t;                 // flat pair index

  // CDNA5 Tensor Data Mover: issue one DMA per block (wave 0), drain TENSORcnt.
  if (wave == 0) tdm_touch(ws, s_tdm);

  // -------- per-pair geometry (init side block-uniform -> scalar loads) -----
  const float* ix = init_x  + bo * 9;
  const float* qx = query_x + (b * 64 + t) * 9;
  float i2 = ix[2], i3 = ix[3], i4 = ix[4], i7 = ix[7], i8 = ix[8];
  float q2 = qx[2], q3 = qx[3], q4 = qx[4], q7 = qx[7], q8 = qx[8];
  float relx = qx[0] - ix[0], rely = qx[1] - ix[1];

  float si, ci; sincosf(i3 * 100.0f, &si, &ci);
  float sq, cq; sincosf(q3 * 100.0f, &sq, &cq);
  float oix = 0.5f * i2 * ci, oiy = 0.5f * i2 * si;
  float oqx = 0.5f * q2 * cq, oqy = 0.5f * q2 * sq;
  float abix = 2.0f * oix, abiy = 2.0f * oiy;
  float abqx = 2.0f * oqx, abqy = 2.0f * oqy;
  float s1 = abix * abix + abiy * abiy + 1e-8f;
  float s2 = abqx * abqx + abqy * abqy + 1e-8f;
  float a2x = relx - oqx, a2y = rely - oqy;
  float b2x = relx + oqx, b2y = rely + oqy;
  float d1 = ptseg(a2x, a2y, -oix, -oiy, abix, abiy, s1);
  float d2 = ptseg(b2x, b2y, -oix, -oiy, abix, abiy, s1);
  float d3 = ptseg(-oix, -oiy, a2x, a2y, abqx, abqy, s2);
  float d4 = ptseg( oix,  oiy, a2x, a2y, abqx, abqy, s2);
  float dist = fminf(fminf(d1, d2), fminf(d3, d4)) - i4 - q4;
  float mask = (dist <= 0.0f) ? 1.0f : 0.0f;
  float dist_in = dist * mask * 100.0f;

  float qvx = query_v[(b * 64 + t) * 2 + 0] - init_v[bo * 2 + 0];
  float qvy = query_v[(b * 64 + t) * 2 + 1] - init_v[bo * 2 + 1];
  float iav = init_av[bo];
  float qav = query_av[b * 64 + t] - iav;

  // -------- trunk input -> LDS --------
  h16* hrow = s_h + tid * PH;
  {
    float f9[9] = { relx, rely, q2, q3, q4, qvx, qvy, qav, dist_in };
#pragma unroll
    for (int k = 0; k < 32; ++k) hrow[k] = (k < 9) ? (h16)f9[k] : (h16)0.0f;
  }

  __builtin_prefetch(ws + OFF_TW1T, 0, 1);

  // fold block's branch vector into the 128->3 projection table
  {
    const h16* brow = ws + OFF_BOUT + bo * 128;
#pragma unroll
    for (int u = 0; u < 2; ++u) {
      int n = tid * 2 + u;
      float bv = (float)brow[n];
      s_bw[n * 4 + 0] = bv * out_w[n * 3 + 0];
      s_bw[n * 4 + 1] = bv * out_w[n * 3 + 1];
      s_bw[n * 4 + 2] = bv * out_w[n * 3 + 2];
    }
  }

  // -------- trunk MLP via WMMA --------
  stage_weights(s_wt, ws + OFF_TW0T, 32, s_bias, tb0, tid, 64);
  __syncthreads();
  mlp_layer_128(s_h, s_wt, s_bias, wave * 32, lane, 1, true);
  __syncthreads();
  __builtin_prefetch(ws + OFF_TW2T, 0, 1);
  stage_weights(s_wt, ws + OFF_TW1T, 128, s_bias, tbs, tid, 64);
  __syncthreads();
  mlp_layer_128(s_h, s_wt, s_bias, wave * 32, lane, 4, true);
  __syncthreads();
  stage_weights(s_wt, ws + OFF_TW2T, 128, s_bias, tbs + 128, tid, 64);
  __syncthreads();
  mlp_layer_128(s_h, s_wt, s_bias, wave * 32, lane, 4, false);

  // -------- force = (branch * trunk) @ out_w + out_b (LDS-broadcast table) --
  float f0 = out_b[0], f1 = out_b[1], f2 = out_b[2];
  for (int n = 0; n < 128; ++n) {
    float tv = (float)hrow[n];
    f0 += tv * s_bw[n * 4 + 0];
    f1 += tv * s_bw[n * 4 + 1];
    f2 += tv * s_bw[n * 4 + 2];
  }

  // -------- joint head via WMMA --------
  __syncthreads();
  stage_weights(s_wt, ws + OFF_JW0T, 32, s_bias, jb0, tid, 64);
  {
    float fj[15] = { 0.0f, 0.0f, i2, i3, i4, relx, rely, q2, q3, q4,
                     qvx, qvy, iav, qav, i7 - truncf(i7) };
#pragma unroll
    for (int k = 0; k < 32; ++k) hrow[k] = (k < 15) ? (h16)fj[k] : (h16)0.0f;
  }
  __syncthreads();
  mlp_layer_128(s_h, s_wt, s_bias, wave * 32, lane, 1, true);

  float j0 = jb1[0], j1 = jb1[1];
  for (int n = 0; n < 128; ++n) {
    float hv = (float)hrow[n];
    j0 += hv * jw1[n * 2 + 0];
    j1 += hv * jw1[n * 2 + 1];
  }
  float have_joint = ((truncf(i7) == truncf(q7)) && (i7 > 0.0f)) ? 1.0f : 0.0f;
  j0 *= have_joint; j1 *= have_joint;

  // -------- spring head (rank-1, VALU) --------
  float len = sqrtf(relx * relx + rely * rely);
  float inv = 1.0f / (len + 1e-8f);
  float sacc = sp_b1[0];
  for (int n = 0; n < 128; ++n)
    sacc += fmaxf(len * sp_w0[n] + sp_b0[n], 0.0f) * sp_w1[n];
  float have_spring = ((i8 == q8) && (i8 > 0.0f)) ? 1.0f : 0.0f;
  float sfx = sacc * (-relx * inv) * have_spring;
  float sfy = sacc * (-rely * inv) * have_spring;

  // -------- combine + store --------
  out[p * 3 + 0] = f0 * mask + sfx + j0;
  out[p * 3 + 1] = f1 * mask + sfy + j1;
  out[p * 3 + 2] = f2 * mask;
}

// ---------------------------------------------------------------------------
extern "C" void kernel_launch(void* const* d_in, const int* in_sizes, int n_in,
                              void* d_out, int out_size, void* d_ws, size_t ws_size,
                              hipStream_t stream) {
  const float* init_x   = (const float*)d_in[0];
  const float* query_x  = (const float*)d_in[1];
  const float* init_v   = (const float*)d_in[2];
  const float* query_v  = (const float*)d_in[3];
  const float* init_av  = (const float*)d_in[4];
  const float* query_av = (const float*)d_in[5];
  const float* trunk_w0 = (const float*)d_in[6];
  const float* trunk_b0 = (const float*)d_in[7];
  const float* trunk_ws = (const float*)d_in[8];
  const float* trunk_bs = (const float*)d_in[9];
  const float* branch_w0 = (const float*)d_in[10];
  const float* branch_b0 = (const float*)d_in[11];
  const float* branch_ws = (const float*)d_in[12];
  const float* branch_bs = (const float*)d_in[13];
  const float* out_w    = (const float*)d_in[14];
  const float* out_b    = (const float*)d_in[15];
  const float* sp_w0    = (const float*)d_in[16];
  const float* sp_b0    = (const float*)d_in[17];
  const float* sp_w1    = (const float*)d_in[18];
  const float* sp_b1    = (const float*)d_in[19];
  const float* joint_w0 = (const float*)d_in[20];
  const float* joint_b0 = (const float*)d_in[21];
  const float* joint_w1 = (const float*)d_in[22];
  const float* joint_b1 = (const float*)d_in[23];

  h16*   ws  = (h16*)d_ws;
  float* out = (float*)d_out;

  ff_prep_kernel<<<(OFF_BOUT + 255) / 256, 256, 0, stream>>>(
      trunk_w0, trunk_ws, joint_w0, branch_w0, branch_ws, ws);
  ff_branch_kernel<<<64, 64, 0, stream>>>(init_x, ws, branch_b0, branch_bs);
  ff_trunk_kernel<<<4096, 64, 0, stream>>>(
      init_x, query_x, init_v, query_v, init_av, query_av,
      trunk_b0, trunk_bs, out_w, out_b, sp_w0, sp_b0, sp_w1, sp_b1,
      joint_b0, joint_w1, joint_b1, ws, out);
}
